// ProtoPNet_16844861735550
// MI455X (gfx1250) — matrix operand order repaired and checked by reading, not compiled, not run on hardware
//
#include <hip/hip_runtime.h>
#include <hip/hip_bf16.h>

// Problem constants (from reference setup_inputs)
#define BATCH 32
#define CH    2048      // K dimension
#define HW    196       // 14*14, M dimension per batch
#define NPROT 2000      // P, N dimension
#define NCLS  200
#define EPS_DIST 1e-7f

#define KC    64        // K-chunk staged in LDS per double-buffer stage
#define MROWS 208       // 13 x 16 padded M rows
#define NKC   (CH / KC) // 32 chunks
#define STG_TRANSFERS (MROWS * (KC / 8))   // 1664 x b128 per chunk

// Workspace layout (bytes)
#define XT_BYTES   (size_t)(BATCH * HW * CH * 2)      // bf16 feats, row-major (b,n,k)
#define PB_BYTES   (size_t)(NPROT * CH * 2)           // bf16 protos, row-major (p,k)
#define X2_BYTES   (size_t)(BATCH * HW * 4)
#define P2_BYTES   (size_t)(NPROT * 4)
#define OFF_XT   ((size_t)0)
#define OFF_PB   (OFF_XT + XT_BYTES)
#define OFF_X2   (OFF_PB + PB_BYTES)
#define OFF_P2   (OFF_X2 + X2_BYTES)
#define OFF_SIM  (OFF_P2 + P2_BYTES)

typedef __attribute__((ext_vector_type(16))) __bf16        v16bf;
typedef __attribute__((ext_vector_type(8)))  float         v8f;
typedef __attribute__((ext_vector_type(8)))  unsigned int  v8u;

__device__ __forceinline__ unsigned short f2bf(float f) {
    // round-to-nearest-even float -> bf16
    unsigned int u = __float_as_uint(f);
    u += 0x7fffu + ((u >> 16) & 1u);
    return (unsigned short)(u >> 16);
}

// ---------------------------------------------------------------------------
// Kernel 1: transpose x (B,C,HW) -> Xt (B,HW,C) with fp32 -> bf16 conversion.
// ---------------------------------------------------------------------------
__global__ __launch_bounds__(256) void k_prep_x(const float* __restrict__ x,
                                                unsigned short* __restrict__ Xt) {
    __shared__ float tile[32][33];
    const int b  = blockIdx.z;
    const int k0 = blockIdx.x * 32;
    const int n0 = blockIdx.y * 32;
    const int tx = threadIdx.x, ty = threadIdx.y;
    #pragma unroll
    for (int j = 0; j < 4; ++j) {
        const int k = k0 + ty + j * 8;
        const int n = n0 + tx;
        float v = 0.0f;
        if (n < HW) v = x[((size_t)b * CH + k) * HW + n];   // coalesced in n
        tile[ty + j * 8][tx] = v;
    }
    __syncthreads();
    #pragma unroll
    for (int j = 0; j < 4; ++j) {
        const int n = n0 + ty + j * 8;
        const int k = k0 + tx;
        if (n < HW)
            Xt[((size_t)b * HW + n) * CH + k] = f2bf(tile[tx][ty + j * 8]);  // coalesced in k
    }
}

// ---------------------------------------------------------------------------
// Kernel 2: X2[b,n] = sum_k x[b,k,n]^2 in fp32.
// ---------------------------------------------------------------------------
__global__ __launch_bounds__(256) void k_x2(const float* __restrict__ x,
                                            float* __restrict__ X2) {
    const int b = blockIdx.x;
    const int n = threadIdx.x;
    if (n >= HW) return;
    float s = 0.0f;
    const float* px = x + (size_t)b * CH * HW + n;
    for (int k = 0; k < CH; ++k) {
        float v = px[(size_t)k * HW];
        s = fmaf(v, v, s);
    }
    X2[b * HW + n] = s;
}

// ---------------------------------------------------------------------------
// Kernel 3: protos fp32 -> bf16 + P2[p] = sum_k protos[p,k]^2.
// ---------------------------------------------------------------------------
__global__ __launch_bounds__(256) void k_prep_p(const float* __restrict__ protos,
                                                unsigned short* __restrict__ Pb,
                                                float* __restrict__ P2) {
    __shared__ float red[256];
    const int p = blockIdx.x;
    float s = 0.0f;
    for (int k = threadIdx.x; k < CH; k += 256) {
        const float v = protos[(size_t)p * CH + k];
        Pb[(size_t)p * CH + k] = f2bf(v);
        s = fmaf(v, v, s);
    }
    red[threadIdx.x] = s;
    __syncthreads();
    for (int st = 128; st > 0; st >>= 1) {
        if (threadIdx.x < st) red[threadIdx.x] += red[threadIdx.x + st];
        __syncthreads();
    }
    if (threadIdx.x == 0) P2[p] = red[0];
}

// ---------------------------------------------------------------------------
// Kernel 4 (main): K-outer fused WMMA GEMM + column-min + sqrt/log.
//  - 13 accumulator tiles (all 208 M rows) live in VGPRs per wave.
//  - A tile (shared by all 8 waves) is async-copied global->LDS, double
//    buffered (ASYNCcnt / s_wait_asynccnt), then read as WMMA A fragments
//    via ds_load_b128 pairs.
//  - B fragments (unique per wave) load straight from global, once per
//    K-step (13x less B traffic than M-outer ordering).
// ---------------------------------------------------------------------------
__global__ __launch_bounds__(256) void k_main(const unsigned short* __restrict__ Xt,
                                              const unsigned short* __restrict__ Pb,
                                              const float* __restrict__ X2,
                                              const float* __restrict__ P2,
                                              float* __restrict__ out_mind,   // (B, NPROT)
                                              float* __restrict__ Sim) {      // (B, NPROT)
    __shared__ unsigned short s_a[2][MROWS * KC];   // 2 x 26624 B
    __shared__ float s_x2[MROWS];

    const int b     = blockIdx.y;
    const int ptile = blockIdx.x;
    const int tid   = threadIdx.x;
    const int wave  = tid >> 5;
    const int lane  = tid & 31;
    const int half  = lane >> 4;
    const int l16   = lane & 15;

    if (tid < MROWS) s_x2[tid] = (tid < HW) ? X2[b * HW + tid] : 1e30f;

    const int p  = ptile * 128 + wave * 16 + l16;     // proto column of this lane
    const int pc = (p < NPROT) ? p : (NPROT - 1);
    // B fragment (32x16 bf16): lane = column p, K = half*16 + {0..15}
    const unsigned short* bbase = Pb + (size_t)pc * CH + half * 16;

    // ------- async stage of one 208xKC A chunk into LDS buffer `buf` -------
    const unsigned short* xbase = Xt + (size_t)b * HW * CH;
    auto stage = [&](int buf, int kc) {
        #pragma unroll
        for (int i = 0; i < (STG_TRANSFERS + 255) / 256; ++i) {
            const int idx = tid + i * 256;
            if (idx < STG_TRANSFERS) {
                const int row  = idx >> 3;            // 0..207
                const int seg  = idx & 7;             // 8 x 16B per row
                const int rowc = (row < HW) ? row : (HW - 1);
                const unsigned long long ga =
                    (unsigned long long)(size_t)(xbase + (size_t)rowc * CH + kc * KC + seg * 8);
                const unsigned int la =
                    (unsigned int)(size_t)(const void*)&s_a[buf][row * KC + seg * 8];
                asm volatile("global_load_async_to_lds_b128 %0, %1, off"
                             :: "v"(la), "v"(ga) : "memory");
            }
        }
    };

    v8f acc[13];
    #pragma unroll
    for (int t = 0; t < 13; ++t) acc[t] = (v8f){};

    stage(0, 0);                                      // prime the pipeline

    for (int kc = 0; kc < NKC; ++kc) {
        asm volatile("s_wait_asynccnt 0x0" ::: "memory");
        __syncthreads();                              // chunk kc visible to all waves
        if (kc + 1 < NKC) stage((kc + 1) & 1, kc + 1);
        const unsigned short* abuf = &s_a[kc & 1][0];

        #pragma unroll
        for (int ks = 0; ks < KC / 32; ++ks) {        // 2 K-steps of 32
            // B fragment: 32 contiguous bytes per lane from global (L2-resident)
            const uint4* pb = (const uint4*)(bbase + kc * KC + ks * 32);
            const uint4  b0 = pb[0];
            const uint4  b1 = pb[1];
            v8u bu = {b0.x, b0.y, b0.z, b0.w, b1.x, b1.y, b1.z, b1.w};
            v16bf bv = __builtin_bit_cast(v16bf, bu);

            #pragma unroll
            for (int t = 0; t < 13; ++t) {
                // A fragment from LDS: row = t*16 + l16, K = ks*32 + half*16 + {0..15}
                const unsigned short* lp = abuf + (t * 16 + l16) * KC + ks * 32 + half * 16;
                const uint4 a0 = *(const uint4*)lp;          // K + {0..7}
                const uint4 a1 = *(const uint4*)(lp + 8);    // K + {8..15}
                v8u au = {a0.x, a0.y, a0.z, a0.w, a1.x, a1.y, a1.z, a1.w};
                v16bf av = __builtin_bit_cast(v16bf, au);
                acc[t] = __builtin_amdgcn_wmma_f32_16x16x32_bf16(
                    false, av, false, bv, (short)0, acc[t], false, false);
            }
        }
        __syncthreads();                              // all reads of buffer done
    }

    // ------- column-min over all 208 rows: C/D lane (half,N), VGPR r -> M=r+8*half
    float runmin = 1e30f;
    #pragma unroll
    for (int t = 0; t < 13; ++t) {
        #pragma unroll
        for (int r = 0; r < 8; ++r) {
            const int mr = t * 16 + r + half * 8;
            runmin = fminf(runmin, s_x2[mr] - 2.0f * acc[t][r]);
        }
    }
    runmin = fminf(runmin, __shfl_xor(runmin, 16, 32));   // fold the two lane-halves

    if (half == 0 && p < NPROT) {
        const float mind2 = runmin + P2[p];
        const float d     = sqrtf(fmaxf(mind2, 1e-12f));
        out_mind[b * NPROT + p] = d;                          // min_distances (flat)
        Sim[b * NPROT + p] = __logf((d + 1.0f) / (d + EPS_DIST));  // max similarity
    }
}

// ---------------------------------------------------------------------------
// Kernel 5: scores[b,c] = sum_p Sim[b,p] * fc_w[c,p].
// ---------------------------------------------------------------------------
__global__ __launch_bounds__(256) void k_scores(const float* __restrict__ Sim,
                                                const float* __restrict__ fcw,
                                                float* __restrict__ scores) {
    const int b = blockIdx.x;
    const int c = threadIdx.x;
    if (c >= NCLS) return;
    float s = 0.0f;
    const float* sp = Sim + (size_t)b * NPROT;
    const float* wp = fcw + (size_t)c * NPROT;
    for (int ppos = 0; ppos < NPROT; ++ppos)
        s = fmaf(sp[ppos], wp[ppos], s);
    scores[b * NCLS + c] = s;
}

// ---------------------------------------------------------------------------
extern "C" void kernel_launch(void* const* d_in, const int* in_sizes, int n_in,
                              void* d_out, int out_size, void* d_ws, size_t ws_size,
                              hipStream_t stream) {
    const float* x      = (const float*)d_in[0];   // (32, 2048, 14, 14)
    const float* protos = (const float*)d_in[1];   // (1, 2000, 2048)
    const float* fcw    = (const float*)d_in[2];   // (200, 2000)
    float* out = (float*)d_out;                    // [scores 32*200 | min_dist 32*2000]

    unsigned char* ws = (unsigned char*)d_ws;
    unsigned short* Xt = (unsigned short*)(ws + OFF_XT);
    unsigned short* Pb = (unsigned short*)(ws + OFF_PB);
    float* X2  = (float*)(ws + OFF_X2);
    float* P2  = (float*)(ws + OFF_P2);
    float* Sim = (float*)(ws + OFF_SIM);

    k_prep_x<<<dim3(CH / 32, (HW + 31) / 32, BATCH), dim3(32, 8), 0, stream>>>(x, Xt);
    k_x2<<<BATCH, 256, 0, stream>>>(x, X2);
    k_prep_p<<<NPROT, 256, 0, stream>>>(protos, Pb, P2);
    k_main<<<dim3(16, BATCH), 256, 0, stream>>>(Xt, Pb, X2, P2,
                                                out + BATCH * NCLS, Sim);
    k_scores<<<BATCH, 256, 0, stream>>>(Sim, fcw, out);
}
